// CrossAttention_27204322853095
// MI455X (gfx1250) — matrix operand order repaired and checked by reading, not compiled, not run on hardware
//
#include <hip/hip_runtime.h>
#include <hip/hip_bf16.h>

#define HEADS   16
#define DHEAD   64
#define DMODEL  1024
#define SEQ     2048

typedef __attribute__((ext_vector_type(16))) _Float16 v16h;
typedef __attribute__((ext_vector_type(8)))  _Float16 v8h;
typedef __attribute__((ext_vector_type(8)))  float    v8f;

static __device__ __forceinline__ v16h pack16(v8h lo, v8h hi) {
  return __builtin_shufflevector(lo, hi, 0,1,2,3,4,5,6,7,8,9,10,11,12,13,14,15);
}

static __device__ __forceinline__ v8f wmma16(v16h a, v16h b, v8f c) {
  // D(16x16 f32) = A(16x32 f16) * B(32x16 f16) + C
  return __builtin_amdgcn_wmma_f32_16x16x32_f16(false, a, false, b, (short)0, c,
                                                false, false);
}

// gfx1250 async global->LDS copy (16B per lane), tracked on ASYNCcnt.
// lds  : wave-relative LDS byte address (low 32 bits of generic pointer)
// goff : per-lane 32-bit byte offset from base
// base : uniform 64-bit global base address (SGPR pair)
static __device__ __forceinline__ void async_copy_b128(unsigned lds, unsigned goff,
                                                       const _Float16* base) {
  asm volatile("global_load_async_to_lds_b128 %0, %1, %2"
               :: "v"(lds), "v"(goff), "s"(base)
               : "memory");
}

static __device__ __forceinline__ unsigned lds_addr(const void* p) {
  return (unsigned)(size_t)p;  // flat->LDS aperture: addr[31:0] is the LDS offset
}

// ---------------------------------------------------------------------------
// f32 -> f16 conversion, 8 elements per thread (b128 in, b128 out)
// ---------------------------------------------------------------------------
__global__ __launch_bounds__(256) void cvt_f32_f16(const float* __restrict__ in,
                                                   _Float16* __restrict__ out,
                                                   int n8) {
  int i = blockIdx.x * 256 + threadIdx.x;
  if (i >= n8) return;
  const float4* p = (const float4*)in;
  float4 a = p[2 * i];
  float4 b = p[2 * i + 1];
  v8h o;
  o[0] = (_Float16)a.x; o[1] = (_Float16)a.y; o[2] = (_Float16)a.z; o[3] = (_Float16)a.w;
  o[4] = (_Float16)b.x; o[5] = (_Float16)b.y; o[6] = (_Float16)b.z; o[7] = (_Float16)b.w;
  *(v8h*)&out[8 * i] = o;
}

// ---------------------------------------------------------------------------
// C[M,N] = A[M,K] * W[N,K]^T   (both f16, row-major over K)
// Block: 256 thr = 8 waves (4 x 2), block tile 128x128, K slab 64.
// Each wave: 32x64 micro-tile = 2x4 WMMA accumulators.
// Double-buffered LDS filled by GLOBAL_LOAD_ASYNC_TO_LDS_B128.
// ---------------------------------------------------------------------------
template <bool F32OUT>
__global__ __launch_bounds__(256) void gemm_nt(const _Float16* __restrict__ A,
                                               const _Float16* __restrict__ W,
                                               _Float16* __restrict__ Ch,
                                               float* __restrict__ Cf,
                                               const float* __restrict__ bias,
                                               int M, int N, int K) {
  __shared__ __align__(16) _Float16 sA[2][128 * 64];
  __shared__ __align__(16) _Float16 sB[2][128 * 64];

  const int t    = threadIdx.x;
  const int lane = t & 31;
  const int half = lane >> 4;
  const int l16  = lane & 15;
  const int w    = t >> 5;       // wave id 0..7
  const int wm   = w & 3;        // wave row (4)
  const int wn   = w >> 2;       // wave col (2)
  const int m0   = blockIdx.x * 128;
  const int n0   = blockIdx.y * 128;

  const int srow = t >> 3;          // 0..31 staging row
  const int scol = (t & 7) * 8;     // 0..56 staging col

  // issue this thread's 8 async copies (4 A-rows + 4 W-rows) for slab kb->buf
  auto stage = [&](int kb, int buf) {
#pragma unroll
    for (int r = 0; r < 4; r++) {
      int rr = r * 32 + srow;
      async_copy_b128(lds_addr(&sA[buf][rr * 64 + scol]),
                      (unsigned)(((size_t)(m0 + rr) * K + kb + scol) * 2), A);
      async_copy_b128(lds_addr(&sB[buf][rr * 64 + scol]),
                      (unsigned)(((size_t)(n0 + rr) * K + kb + scol) * 2), W);
    }
  };

  v8f acc[2][4];
#pragma unroll
  for (int a = 0; a < 2; a++)
#pragma unroll
    for (int b = 0; b < 4; b++)
#pragma unroll
      for (int i = 0; i < 8; i++) acc[a][b][i] = 0.0f;

  stage(0, 0);

  int buf = 0;
  for (int kb = 0; kb < K; kb += 64) {
    if (kb + 64 < K) {
      stage(kb + 64, buf ^ 1);
      // 8 newer copies outstanding; in-order completion => current buf done
      asm volatile("s_wait_asynccnt 0x8" ::: "memory");
    } else {
      asm volatile("s_wait_asynccnt 0x0" ::: "memory");
    }
    __syncthreads();

#pragma unroll
    for (int ks = 0; ks < 2; ks++) {
      const int ko = ks * 32;
      v16h af[2], bf[4];
#pragma unroll
      for (int mi = 0; mi < 2; mi++) {
        int row = wm * 32 + mi * 16 + l16;
        v8h lo = *(const v8h*)&sA[buf][row * 64 + ko + half * 8];
        v8h hi = *(const v8h*)&sA[buf][row * 64 + ko + 16 + half * 8];
        af[mi] = pack16(lo, hi);
      }
#pragma unroll
      for (int ni = 0; ni < 4; ni++) {
        int col = wn * 64 + ni * 16 + l16;
        v8h lo = *(const v8h*)&sB[buf][col * 64 + ko + half * 8];
        v8h hi = *(const v8h*)&sB[buf][col * 64 + ko + 16 + half * 8];
        bf[ni] = pack16(lo, hi);
      }
#pragma unroll
      for (int mi = 0; mi < 2; mi++)
#pragma unroll
        for (int ni = 0; ni < 4; ni++)
          acc[mi][ni] = wmma16(af[mi], bf[ni], acc[mi][ni]);
    }
    __syncthreads();
    buf ^= 1;
  }

  // ---- epilogue: lane = col n (l16), vgpr i -> row 8*half+i ----
#pragma unroll
  for (int mi = 0; mi < 2; mi++)
#pragma unroll
    for (int ni = 0; ni < 4; ni++)
#pragma unroll
      for (int i = 0; i < 8; i++) {
        int gm = m0 + wm * 32 + mi * 16 + half * 8 + i;
        int gn = n0 + wn * 64 + ni * 16 + l16;
        if constexpr (F32OUT)
          Cf[(size_t)gm * N + gn] = acc[mi][ni][i] + bias[gn];
        else
          Ch[(size_t)gm * N + gn] = (_Float16)acc[mi][ni][i];
      }
}

// ---------------------------------------------------------------------------
// Flash attention per (batch, head), TRANSPOSED score layout:
//   S^T = K * Q^T  (M=keys, N=queries)  -> lane owns one query column
//   O'  = V^T * P  (M=head-dim, N=queries)
// grid.x = B*HEADS, grid.y = SEQ/64, block = 128 thr (4 waves, 16 q each)
// K tile staged by async-to-LDS; V staged transposed via packed b32 stores.
// ---------------------------------------------------------------------------
__global__ __launch_bounds__(128) void attn_kernel(const _Float16* __restrict__ Q,
                                                   const _Float16* __restrict__ Kp,
                                                   const _Float16* __restrict__ Vp,
                                                   _Float16* __restrict__ O,
                                                   int S) {
  __shared__ __align__(16) _Float16 sK[64 * 64];     // [key][d]
  __shared__ __align__(16) _Float16 sV[64 * 64];     // transposed [d][key]
  __shared__ __align__(16) _Float16 sP[4][16 * 64];  // per-wave [query][key]

  const int t    = threadIdx.x;
  const int wid  = t >> 5;
  const int lane = t & 31;
  const int half = lane >> 4;
  const int l16  = lane & 15;

  const int bh = blockIdx.x;
  const int b  = bh / HEADS;
  const int h  = bh % HEADS;
  const int n0 = blockIdx.y * 64;
  const size_t rowBase = (size_t)b * S;
  const int colBase    = h * DHEAD;
  const int r0 = wid * 16;  // this wave's query sub-block

  // ---- Q as B-fragments of K*Q^T, loaded straight from global ----
  // lane column n = query r0+l16; element j -> d = f*32 + half*16 + j
  // fold softmax scale 1/sqrt(64) = 0.125 into Q (exact in f16)
  v16h qb[2];
  {
    const size_t qrow = rowBase + n0 + r0 + l16;
#pragma unroll
    for (int f = 0; f < 2; f++) {
      v8h lo = *(const v8h*)&Q[qrow * DMODEL + colBase + f * 32 + half * 16];
      v8h hi = *(const v8h*)&Q[qrow * DMODEL + colBase + f * 32 + half * 16 + 8];
      v16h q = pack16(lo, hi);
#pragma unroll
      for (int j = 0; j < 16; j++) q[j] = q[j] * (_Float16)0.125f;
      qb[f] = q;
    }
  }

  // O' accumulators: o[mt] covers d rows mt*16..+16, cols = 16 queries
  v8f o[4];
#pragma unroll
  for (int mt = 0; mt < 4; mt++)
#pragma unroll
    for (int i = 0; i < 8; i++) o[mt][i] = 0.0f;

  float rm = -__builtin_inff();  // running max for this lane's query
  float rs = 0.0f;               // running denominator

  const int nTiles = S / 64;
  for (int kt = 0; kt < nTiles; kt++) {
    // ---- stage K row-major [key][d] via async copies ----
    {
      const int row = t >> 3;        // 0..15
      const int col = (t & 7) * 8;
#pragma unroll
      for (int r = 0; r < 4; r++) {
        int rr = r * 16 + row;
        async_copy_b128(lds_addr(&sK[rr * 64 + col]),
                        (unsigned)(((rowBase + kt * 64 + rr) * DMODEL + colBase + col) * 2),
                        Kp);
      }
    }
    // ---- stage V transposed [d][key], packed key-pair b32 stores ----
    {
      const int kp2 = (t & 31) * 2;   // keys kp2, kp2+1
      const int dg0 = t >> 5;         // 0..3
#pragma unroll
      for (int r = 0; r < 2; r++) {
        int dg = r * 4 + dg0;         // d group 0..7 -> d = dg*8..+8
        v8h a = *(const v8h*)&Vp[(rowBase + kt * 64 + kp2) * DMODEL + colBase + dg * 8];
        v8h c = *(const v8h*)&Vp[(rowBase + kt * 64 + kp2 + 1) * DMODEL + colBase + dg * 8];
#pragma unroll
        for (int j = 0; j < 8; j++) {
          union { _Float16 hh[2]; unsigned u; } pk;
          pk.hh[0] = a[j];
          pk.hh[1] = c[j];
          *(unsigned*)&sV[(dg * 8 + j) * 64 + kp2] = pk.u;
        }
      }
    }
    asm volatile("s_wait_asynccnt 0x0" ::: "memory");
    __syncthreads();

    // ---- S^T tiles: sc[mt] = K[mt*16..+16] x Qs^T  (16 keys x 16 queries) ----
    v8f sc[4];
#pragma unroll
    for (int mt = 0; mt < 4; mt++) {
      v8f a;
#pragma unroll
      for (int i = 0; i < 8; i++) a[i] = 0.0f;
#pragma unroll
      for (int f = 0; f < 2; f++) {
        int krow = mt * 16 + l16;
        v8h lo = *(const v8h*)&sK[krow * 64 + f * 32 + half * 16];
        v8h hi = *(const v8h*)&sK[krow * 64 + f * 32 + half * 16 + 8];
        a = wmma16(pack16(lo, hi), qb[f], a);
      }
      sc[mt] = a;
    }

    // ---- online softmax: keys are in-lane (+ one xor-16 exchange) ----
    float m = sc[0][0];
#pragma unroll
    for (int mt = 0; mt < 4; mt++)
#pragma unroll
      for (int i = 0; i < 8; i++) m = fmaxf(m, sc[mt][i]);
    m = fmaxf(m, __shfl_xor(m, 16, 32));

    float nm   = fmaxf(rm, m);
    float scal = __expf(rm - nm);

    float s = 0.0f;
#pragma unroll
    for (int mt = 0; mt < 4; mt++)
#pragma unroll
      for (int i = 0; i < 8; i++) {
        float p = __expf(sc[mt][i] - nm);
        sc[mt][i] = p;
        s += p;
      }
    s += __shfl_xor(s, 16, 32);

    rs = rs * scal + s;
    rm = nm;
#pragma unroll
    for (int mt = 0; mt < 4; mt++)
#pragma unroll
      for (int i = 0; i < 8; i++) o[mt][i] *= scal;

    // ---- P -> per-wave LDS [query][key], b128 stores (keys contiguous) ----
#pragma unroll
    for (int mt = 0; mt < 4; mt++) {
      v8h pv;
#pragma unroll
      for (int i = 0; i < 8; i++) pv[i] = (_Float16)sc[mt][i];
      *(v8h*)&sP[wid][l16 * 64 + mt * 16 + half * 8] = pv;
    }

    // ---- O' += V^T * P ----
    v16h pb[2];
#pragma unroll
    for (int f = 0; f < 2; f++) {
      v8h lo = *(const v8h*)&sP[wid][l16 * 64 + f * 32 + half * 16];
      v8h hi = *(const v8h*)&sP[wid][l16 * 64 + f * 32 + half * 16 + 8];
      pb[f] = pack16(lo, hi);
    }
#pragma unroll
    for (int mt = 0; mt < 4; mt++) {
#pragma unroll
      for (int f = 0; f < 2; f++) {
        int drow = mt * 16 + l16;
        v8h lo = *(const v8h*)&sV[drow * 64 + f * 32 + half * 16];
        v8h hi = *(const v8h*)&sV[drow * 64 + f * 32 + half * 16 + 8];
        o[mt] = wmma16(pack16(lo, hi), pb[f], o[mt]);
      }
    }
    __syncthreads();
  }

  // ---- normalize, write: lane owns query row; d packs into b128 stores ----
  {
    const float inv = 1.0f / rs;
    const size_t orow = rowBase + n0 + r0 + l16;
#pragma unroll
    for (int mt = 0; mt < 4; mt++) {
      v8h ov;
#pragma unroll
      for (int i = 0; i < 8; i++) ov[i] = (_Float16)(o[mt][i] * inv);
      *(v8h*)&O[orow * DMODEL + colBase + mt * 16 + half * 8] = ov;
    }
  }
}

// ---------------------------------------------------------------------------
extern "C" void kernel_launch(void* const* d_in, const int* in_sizes, int n_in,
                              void* d_out, int out_size, void* d_ws, size_t ws_size,
                              hipStream_t stream) {
  const float* x   = (const float*)d_in[0];
  const float* ctx = (const float*)d_in[1];
  const float* Wq  = (const float*)d_in[2];
  const float* Wk  = (const float*)d_in[3];
  const float* Wv  = (const float*)d_in[4];
  const float* Wo  = (const float*)d_in[5];
  const float* bo  = (const float*)d_in[6];

  const int Bn = in_sizes[0] / (SEQ * DMODEL);  // batch = 2
  const int M  = Bn * SEQ;                      // 4096 flattened rows

  _Float16* ws = (_Float16*)d_ws;
  size_t off = 0;
  _Float16* xh  = ws + off; off += (size_t)M * DMODEL;
  _Float16* cxh = ws + off; off += (size_t)M * DMODEL;
  _Float16* wqh = ws + off; off += (size_t)DMODEL * DMODEL;
  _Float16* wkh = ws + off; off += (size_t)DMODEL * DMODEL;
  _Float16* wvh = ws + off; off += (size_t)DMODEL * DMODEL;
  _Float16* woh = ws + off; off += (size_t)DMODEL * DMODEL;
  _Float16* Qp  = ws + off; off += (size_t)M * DMODEL;
  _Float16* Kp  = ws + off; off += (size_t)M * DMODEL;
  _Float16* Vp  = ws + off; off += (size_t)M * DMODEL;
  _Float16* Ao  = ws + off; off += (size_t)M * DMODEL;

  // 1) f32 -> f16 conversions
  {
    struct { const float* s; _Float16* d; size_t n; } jobs[6] = {
        {x,  xh,  (size_t)M * DMODEL}, {ctx, cxh, (size_t)M * DMODEL},
        {Wq, wqh, (size_t)DMODEL * DMODEL}, {Wk, wkh, (size_t)DMODEL * DMODEL},
        {Wv, wvh, (size_t)DMODEL * DMODEL}, {Wo, woh, (size_t)DMODEL * DMODEL}};
    for (int j = 0; j < 6; j++) {
      int n8 = (int)(jobs[j].n / 8);
      cvt_f32_f16<<<(n8 + 255) / 256, 256, 0, stream>>>(jobs[j].s, jobs[j].d, n8);
    }
  }

  // 2) Q/K/V projections (f16 GEMMs), block tile 128x128, async double-buffer
  dim3 gg(M / 128, DMODEL / 128);
  gemm_nt<false><<<gg, 256, 0, stream>>>(xh,  wqh, Qp, nullptr, nullptr, M, DMODEL, DMODEL);
  gemm_nt<false><<<gg, 256, 0, stream>>>(cxh, wkh, Kp, nullptr, nullptr, M, DMODEL, DMODEL);
  gemm_nt<false><<<gg, 256, 0, stream>>>(cxh, wvh, Vp, nullptr, nullptr, M, DMODEL, DMODEL);

  // 3) flash attention per (b,h)
  dim3 ga(Bn * HEADS, SEQ / 64);
  attn_kernel<<<ga, 128, 0, stream>>>(Qp, Kp, Vp, Ao, SEQ);

  // 4) output projection (f32 out + bias)
  gemm_nt<true><<<gg, 256, 0, stream>>>(Ao, woh, nullptr, (float*)d_out, bo, M,
                                        DMODEL, DMODEL);
}
